// Attention_24498493456983
// MI455X (gfx1250) — matrix operand order repaired and checked by reading, not compiled, not run on hardware
//
#include <hip/hip_runtime.h>
#include <hip/hip_bf16.h>

// ---------------------------------------------------------------------------
// Multi-head attention forward for MI455X (gfx1250), bf16 WMMA pipeline.
// Shapes fixed by the reference: B=4, S=2048, E=1024, H=16, D=64.
// ---------------------------------------------------------------------------

typedef __bf16 v16bf __attribute__((ext_vector_type(16)));
typedef __bf16 v8bf  __attribute__((ext_vector_type(8)));
typedef float  v8f   __attribute__((ext_vector_type(8)));

#define BB 4
#define SS 2048
#define EE 1024
#define HH 16
#define DD 64

__device__ __forceinline__ v16bf pack16(v8bf lo, v8bf hi) {
  return __builtin_shufflevector(lo, hi, 0,1,2,3,4,5,6,7,8,9,10,11,12,13,14,15);
}

__device__ __forceinline__ v8f wmma_bf16(v16bf a, v16bf b, v8f c) {
  // D(f32 16x16) = A(bf16 16x32) * B(bf16 32x16) + C
  return __builtin_amdgcn_wmma_f32_16x16x32_bf16(false, a, false, b,
                                                 (short)0, c, false, false);
}

// --------------------------- conversion kernels ----------------------------

__global__ void k_cvt_bf16(const float* __restrict__ in, __bf16* __restrict__ out, int n) {
  int i = blockIdx.x * 256 + threadIdx.x;
  if (i < n) out[i] = (__bf16)in[i];
}

// W_{Q,K,V}: [H,E,D] f32 -> [H,D,E] bf16 (transposed so B-operand K-dim is contiguous)
__global__ void k_tr_wqkv(const float* __restrict__ in, __bf16* __restrict__ out) {
  int idx = blockIdx.x * 256 + threadIdx.x;      // over H*D*E = 1M
  int e = idx & (EE - 1);
  int d = (idx >> 10) & (DD - 1);
  int h = idx >> 16;
  out[idx] = (__bf16)in[(size_t)h * EE * DD + (size_t)e * DD + d];
}

// W_O: [H,D,E] f32 (== [K=H*D, E] row-major) -> [E, H*D] bf16
__global__ void k_tr_wo(const float* __restrict__ in, __bf16* __restrict__ out) {
  int idx = blockIdx.x * 256 + threadIdx.x;      // out index = e*1024 + k
  int k = idx & 1023;
  int e = idx >> 10;
  out[idx] = (__bf16)in[(size_t)k * EE + e];
}

// ---------------- shared GEMM inner machinery (16x64 per wave) -------------

struct OpBuf { v16bf a; v16bf b[4]; };

// Load one k-step's operands: A tile row chunk + 4 B tiles (ldb = K stride).
__device__ __forceinline__ OpBuf load_ops(const __bf16* Ar, const __bf16* Bw,
                                          size_t ldb, int k) {
  OpBuf o;
  o.a = pack16(*(const v8bf*)(Ar + k), *(const v8bf*)(Ar + k + 16));
#pragma unroll
  for (int nt = 0; nt < 4; ++nt) {
    const __bf16* Br = Bw + (size_t)(nt * 16) * ldb + k;
    o.b[nt] = pack16(*(const v8bf*)Br, *(const v8bf*)(Br + 16));
  }
  return o;
}

__device__ __forceinline__ void consume(v8f acc[4], const OpBuf& o) {
#pragma unroll
  for (int nt = 0; nt < 4; ++nt) acc[nt] = wmma_bf16(o.a, o.b[nt], acc[nt]);
}

// ------------------------------ QKV GEMM -----------------------------------
// Per (b,h): out[S,D] = x[b] (S x E) @ Wt[h]^T, Wt stored [D,E].
// One wave per 16x64 tile (4 accumulators). Ping-pong double-buffered k-loop:
// two operand sets alternate so loads for step k+32 are in flight while the
// WMMAs for step k issue, with no rotation copies between iterations.
// transposed=1 stores V as [D,S] (so PV B-operand reads are contiguous).
__global__ __launch_bounds__(256)
void k_qkv_gemm(const __bf16* __restrict__ xb, const __bf16* __restrict__ Wt,
                __bf16* __restrict__ out, float scale, int transposed) {
  const int wid = threadIdx.x >> 5, lane = threadIdx.x & 31;
  const int ln = lane & 15, hf = lane >> 4;
  const int t  = blockIdx.x * 8 + wid;           // 8192 wave tiles (16x64 each)
  const int mt = t & 127;                        // S tile (128)
  const int bh = t >> 7;                         // b*16+h (64)
  const int b  = bh >> 4, h = bh & 15;

  const __bf16* Ar = xb + ((size_t)(b * SS + mt * 16 + ln)) * EE + hf * 8;
  const __bf16* Bw = Wt + (size_t)h * DD * EE + (size_t)ln * EE + hf * 8;

  v8f acc[4] = {};
  OpBuf p0 = load_ops(Ar, Bw, EE, 0);
  OpBuf p1;
  for (int k = 0; k < EE - 64; k += 64) {
    p1 = load_ops(Ar, Bw, EE, k + 32);
    consume(acc, p0);                            // step k
    p0 = load_ops(Ar, Bw, EE, k + 64);
    consume(acc, p1);                            // step k+32
  }
  p1 = load_ops(Ar, Bw, EE, EE - 32);
  consume(acc, p0);                              // step EE-64
  consume(acc, p1);                              // step EE-32

#pragma unroll
  for (int nt = 0; nt < 4; ++nt)
#pragma unroll
    for (int r = 0; r < 8; ++r) {
      int srow = mt * 16 + r + 8 * hf;
      int col  = nt * 16 + ln;
      float v  = acc[nt][r] * scale;
      if (transposed)
        out[(size_t)bh * DD * SS + (size_t)col * SS + srow] = (__bf16)v;
      else
        out[(size_t)bh * SS * DD + (size_t)srow * DD + col] = (__bf16)v;
    }
}

// ----------------------------- attention -----------------------------------

__device__ __forceinline__ v8f score16(const __bf16* Kbase, int k0,
                                       v16bf aq0, v16bf aq1, int ln, int hf) {
  const __bf16* kr = Kbase + (size_t)(k0 + ln) * DD + hf * 8;
  v16bf b0 = pack16(*(const v8bf*)kr,        *(const v8bf*)(kr + 16));
  v16bf b1 = pack16(*(const v8bf*)(kr + 32), *(const v8bf*)(kr + 48));
  v8f c = {};
  c = wmma_bf16(aq0, b0, c);   // d = 0..31
  c = wmma_bf16(aq1, b1, c);   // d = 32..63
  return c;
}

// Workgroup: 16 waves (512 thr) = one (b, 16-row q tile); wave w = head w.
// Dynamic LDS: 16*2048 f32 attn-mean accumulator (128KB) + 16*(16x32) bf16 stage.
__global__ __launch_bounds__(512)
void k_attn(const __bf16* __restrict__ Qb, const __bf16* __restrict__ Kb,
            const __bf16* __restrict__ Vt, __bf16* __restrict__ zb,
            float* __restrict__ attn_out, const int* __restrict__ flag) {
  extern __shared__ char dynsmem[];
  float*  lds_attn = (float*)dynsmem;                       // 16 rows x 2048
  __bf16* stage    = (__bf16*)(dynsmem + 16 * SS * 4);      // 16 waves x 16x32

  const int tid = threadIdx.x, wid = tid >> 5, lane = tid & 31;
  const int ln = lane & 15, hf = lane >> 4;
  const int b = blockIdx.x >> 7, qt = blockIdx.x & 127;
  const int q0 = qt * 16, bh = b * HH + wid;
  const int causal = *flag;

  // zero head-mean accumulator (harness poisons memory)
  float4* az = (float4*)lds_attn;
  for (int i = tid; i < 16 * SS / 4; i += 512) az[i] = make_float4(0.f, 0.f, 0.f, 0.f);
  __syncthreads();

  // Q tile as two A-operands (pre-scaled by 1/sqrt(D) at projection time)
  const __bf16* qr = Qb + (size_t)bh * SS * DD + (size_t)(q0 + ln) * DD + hf * 8;
  v16bf aq0 = pack16(*(const v8bf*)qr,        *(const v8bf*)(qr + 16));
  v16bf aq1 = pack16(*(const v8bf*)(qr + 32), *(const v8bf*)(qr + 48));
  const __bf16* Kbase = Kb + (size_t)bh * SS * DD;
  const __bf16* Vbase = Vt + (size_t)bh * DD * SS;   // [D, S]

  const int ktEnd = causal ? qt : (SS / 16 - 1);

  // ---- pass 1: streaming row max / sum (flash-style, WMMA scores) ----
  float m[8], s[8];
#pragma unroll
  for (int r = 0; r < 8; ++r) { m[r] = -__builtin_inff(); s[r] = 0.f; }

  for (int kt = 0; kt <= ktEnd; ++kt) {
    v8f c = score16(Kbase, kt * 16, aq0, aq1, ln, hf);
    if (causal && kt == qt) {
#pragma unroll
      for (int r = 0; r < 8; ++r) if (ln > r + 8 * hf) c[r] = -100000.f;
    }
#pragma unroll
    for (int r = 0; r < 8; ++r) {
      float tm = c[r];                               // row lives across 16 lanes
      tm = fmaxf(tm, __shfl_xor(tm, 1));
      tm = fmaxf(tm, __shfl_xor(tm, 2));
      tm = fmaxf(tm, __shfl_xor(tm, 4));
      tm = fmaxf(tm, __shfl_xor(tm, 8));
      float mn = fmaxf(m[r], tm);
      float e  = __expf(c[r] - mn);
      float ts = e;
      ts += __shfl_xor(ts, 1);
      ts += __shfl_xor(ts, 2);
      ts += __shfl_xor(ts, 4);
      ts += __shfl_xor(ts, 8);
      s[r] = s[r] * __expf(m[r] - mn) + ts;
      m[r] = mn;
    }
  }
  float inv_s[8];
#pragma unroll
  for (int r = 0; r < 8; ++r) inv_s[r] = 1.0f / s[r];

  // ---- pass 2: probs -> head-mean (LDS atomics) + PV (WMMA) ----
  v8f acc[4] = {};                                   // z tile 16 x 64 (f32)
  __bf16* st = stage + wid * 512;                    // private 16x32 staging
  const int nkt = ktEnd + 1;

  for (int kp = 0; kp < nkt; kp += 2) {
#pragma unroll
    for (int h2 = 0; h2 < 2; ++h2) {
      const int kt = kp + h2;
      if (kt < nkt) {
        v8f c = score16(Kbase, kt * 16, aq0, aq1, ln, hf);
        if (causal && kt == qt) {
#pragma unroll
          for (int r = 0; r < 8; ++r) if (ln > r + 8 * hf) c[r] = -100000.f;
        }
#pragma unroll
        for (int r = 0; r < 8; ++r) {
          float p = __expf(c[r] - m[r]) * inv_s[r];
          atomicAdd(&lds_attn[(r + 8 * hf) * SS + kt * 16 + ln], p * 0.0625f);
          st[(r + 8 * hf) * 32 + h2 * 16 + ln] = (__bf16)p;   // transpose stage
        }
      } else {                                       // odd tail: zero half-tile
#pragma unroll
        for (int r = 0; r < 8; ++r) st[(r + 8 * hf) * 32 + 16 + ln] = (__bf16)0.f;
      }
    }
    // same-wave LDS store->load ordering (hardware in-order; explicit wait)
    asm volatile("s_wait_dscnt 0x0" ::: "memory");
    v16bf ap = pack16(*(const v8bf*)&st[ln * 32 + hf * 8],
                      *(const v8bf*)&st[ln * 32 + hf * 8 + 16]);
#pragma unroll
    for (int n = 0; n < 4; ++n) {
      const __bf16* vr = Vbase + (size_t)(n * 16 + ln) * SS + kp * 16 + hf * 8;
      v16bf bv = pack16(*(const v8bf*)vr, *(const v8bf*)(vr + 16));
      acc[n] = wmma_bf16(ap, bv, acc[n]);
    }
  }

  // store z as [B, S, H, D] bf16 so output projection is one flat GEMM
#pragma unroll
  for (int n = 0; n < 4; ++n)
#pragma unroll
    for (int r = 0; r < 8; ++r) {
      int q = q0 + r + 8 * hf;
      zb[(((size_t)b * SS + q) * HH + wid) * DD + n * 16 + ln] = (__bf16)acc[n][r];
    }

  __syncthreads();                                   // all heads accumulated
  float* oa = attn_out + ((size_t)b * SS + q0) * SS; // 16 rows, coalesced f128
  for (int i = tid; i < 16 * SS / 4; i += 512) ((float4*)oa)[i] = az[i];
}

// --------------------------- output projection -----------------------------
// o[B*S, E] = zb[B*S, H*D] @ Wflat[H*D, E];  WoT stored [E, H*D].
// One wave per 16x64 tile, ping-pong double-buffered k-loop (same as QKV).
__global__ __launch_bounds__(256)
void k_oproj(const __bf16* __restrict__ zb, const __bf16* __restrict__ WoT,
             float* __restrict__ o) {
  const int wid = threadIdx.x >> 5, lane = threadIdx.x & 31;
  const int ln = lane & 15, hf = lane >> 4;
  const int t  = blockIdx.x * 8 + wid;               // 8192 wave tiles (16x64)
  const int g  = t & 15, mt = t >> 4;                // g: group of 4 N-tiles
  const int KD = HH * DD;                            // 1024

  const __bf16* Ar = zb  + (size_t)(mt * 16 + ln) * KD + hf * 8;
  const __bf16* Bw = WoT + (size_t)(g * 64 + ln) * KD + hf * 8;

  v8f acc[4] = {};
  OpBuf p0 = load_ops(Ar, Bw, KD, 0);
  OpBuf p1;
  for (int k = 0; k < KD - 64; k += 64) {
    p1 = load_ops(Ar, Bw, KD, k + 32);
    consume(acc, p0);
    p0 = load_ops(Ar, Bw, KD, k + 64);
    consume(acc, p1);
  }
  p1 = load_ops(Ar, Bw, KD, KD - 32);
  consume(acc, p0);
  consume(acc, p1);

#pragma unroll
  for (int n = 0; n < 4; ++n)
#pragma unroll
    for (int r = 0; r < 8; ++r)
      o[(size_t)(mt * 16 + r + 8 * hf) * EE + g * 64 + n * 16 + ln] = acc[n][r];
}

// ------------------------------- launcher ----------------------------------

extern "C" void kernel_launch(void* const* d_in, const int* in_sizes, int n_in,
                              void* d_out, int out_size, void* d_ws, size_t ws_size,
                              hipStream_t stream) {
  (void)in_sizes; (void)n_in; (void)out_size; (void)ws_size;
  const float* x  = (const float*)d_in[0];
  const float* WQ = (const float*)d_in[1];
  const float* WK = (const float*)d_in[2];
  const float* WV = (const float*)d_in[3];
  const float* WO = (const float*)d_in[4];
  const int* flag = (const int*)d_in[5];

  char* ws = (char*)d_ws;
  const size_t NX = (size_t)BB * SS * EE;            // 8388608
  const size_t NP = (size_t)BB * HH * SS * DD;       // 8388608
  const size_t NW = (size_t)HH * DD * EE;            // 1048576
  size_t off = 0;
  __bf16* xb  = (__bf16*)(ws + off); off += NX * 2;
  __bf16* Qb  = (__bf16*)(ws + off); off += NP * 2;
  __bf16* Kb  = (__bf16*)(ws + off); off += NP * 2;
  __bf16* Vt  = (__bf16*)(ws + off); off += NP * 2;
  __bf16* zb  = (__bf16*)(ws + off); off += NP * 2;
  __bf16* WtQ = (__bf16*)(ws + off); off += NW * 2;
  __bf16* WtK = (__bf16*)(ws + off); off += NW * 2;
  __bf16* WtV = (__bf16*)(ws + off); off += NW * 2;
  __bf16* WoT = (__bf16*)(ws + off); off += NW * 2;  // total ~88MB

  k_cvt_bf16<<<(int)(NX / 256), 256, 0, stream>>>(x, xb, (int)NX);
  k_tr_wqkv <<<(int)(NW / 256), 256, 0, stream>>>(WQ, WtQ);
  k_tr_wqkv <<<(int)(NW / 256), 256, 0, stream>>>(WK, WtK);
  k_tr_wqkv <<<(int)(NW / 256), 256, 0, stream>>>(WV, WtV);
  k_tr_wo   <<<(int)(NW / 256), 256, 0, stream>>>(WO, WoT);

  k_qkv_gemm<<<1024, 256, 0, stream>>>(xb, WtQ, Qb, 0.125f, 0);  // Q * 1/sqrt(64)
  k_qkv_gemm<<<1024, 256, 0, stream>>>(xb, WtK, Kb, 1.0f,   0);
  k_qkv_gemm<<<1024, 256, 0, stream>>>(xb, WtV, Vt, 1.0f,   1);  // V transposed

  float* o    = (float*)d_out;
  float* attn = o + NX;                              // outputs concatenated
  const int smem = 16 * SS * 4 + HH * 16 * 32 * 2;   // 128KB + 16KB
  k_attn<<<BB * (SS / 16), 512, smem, stream>>>(Qb, Kb, Vt, zb, attn, flag);

  k_oproj<<<1024, 256, 0, stream>>>(zb, WoT, o);
}
